// MultiHeadAttention_28286654611814
// MI455X (gfx1250) — matrix operand order repaired and checked by reading, not compiled, not run on hardware
//
#include <hip/hip_runtime.h>
#include <hip/hip_bf16.h>

// ---------------------------------------------------------------------------
// MI455X (gfx1250) multi-head attention forward.
// wave32, WMMA f16 (f32 accum) for all GEMMs, flash-style online softmax.
// ---------------------------------------------------------------------------

typedef __attribute__((ext_vector_type(16))) _Float16 v16h;
typedef __attribute__((ext_vector_type(8)))  _Float16 v8h;
typedef __attribute__((ext_vector_type(8)))  float    v8f;

union AccF { v8f v; float f[8]; };

#define B_   2
#define N_   2048
#define D_   1024
#define H_   16
#define DH_  64
#define BN_  (B_ * N_)

__device__ __forceinline__ v16h make_v16(v8h lo, v8h hi) {
  return __builtin_shufflevector(lo, hi, 0,1,2,3,4,5,6,7,8,9,10,11,12,13,14,15);
}

// A-matrix fragment (16x32, f16). Element (m,k) = src[(row0+m)*ld + k0 + k].
// ISA layout: lanes 0-15: m=lane, k 0-7 in elems 0-7, k 16-23 in elems 8-15.
//             lanes 16-31: m=lane-16, k 8-15 / 24-31.
__device__ __forceinline__ v16h load_a_frag(const _Float16* p, int ld, int row0, int k0) {
  int lane = threadIdx.x & 31;
  int m    = lane & 15;
  int klo  = (lane < 16) ? 0 : 8;
  const _Float16* r = p + (size_t)(row0 + m) * ld + k0;
  v8h lo = *(const v8h*)(r + klo);        // 16B load
  v8h hi = *(const v8h*)(r + klo + 16);   // 16B load
  return make_v16(lo, hi);
}

// B-matrix fragment (32x16, f16). Element (k,n) = src[(n0+n)*ld + k0 + k]
// (i.e. the source is "transposed": contiguous along k).
// ISA layout: lanes 0-15: n=lane, k 0-7 / 16-23; lanes 16-31: n=lane-16, k 8-15 / 24-31.
__device__ __forceinline__ v16h load_bt_frag(const _Float16* p, int ld, int n0, int k0) {
  int lane = threadIdx.x & 31;
  int n    = n0 + (lane & 15);
  int kk   = k0 + ((lane < 16) ? 0 : 8);
  const _Float16* r = p + (size_t)n * ld + kk;
  v8h lo = *(const v8h*)(r);              // 16B load
  v8h hi = *(const v8h*)(r + 16);         // 16B load
  return make_v16(lo, hi);
}

__device__ __forceinline__ v8f wmma_f16(v16h a, v16h b, v8f c) {
  // D = A(16x32) * B(32x16) + C, f32 accum
  return __builtin_amdgcn_wmma_f32_16x16x32_f16(false, a, false, b, (short)0, c,
                                                false, false);
}

// ---------------------------------------------------------------------------
// Stage 0: fp32 -> f16 conversion
// ---------------------------------------------------------------------------
__global__ void cvt_f32_f16(const float* __restrict__ src,
                            _Float16* __restrict__ dst, int n) {
  int i = blockIdx.x * blockDim.x + threadIdx.x;
  int stride = gridDim.x * blockDim.x;
  for (; i < n; i += stride) dst[i] = (_Float16)src[i];
}

// ---------------------------------------------------------------------------
// Stage 1: fused QKV projection + RoPE (Q,K) + transpose-store (V).
// grid: (BN/128, H, 3)  block: 256 (8 waves, each a 16x64 tile = one head)
// ---------------------------------------------------------------------------
__global__ __launch_bounds__(256)
void qkv_proj(const _Float16* __restrict__ xh,
              const _Float16* __restrict__ Wq,
              const _Float16* __restrict__ Wk,
              const _Float16* __restrict__ Wv,
              _Float16* __restrict__ Qh,   // [B,H,N,dh]
              _Float16* __restrict__ Kh,   // [B,H,N,dh]
              _Float16* __restrict__ Vt) { // [B,H,dh,N] (transposed)
  int wave = threadIdx.x >> 5;
  int lane = threadIdx.x & 31;
  int mt    = blockIdx.x * 128 + wave * 16;   // row tile within [0, BN)
  int head  = blockIdx.y;
  int which = blockIdx.z;                     // 0=Q 1=K 2=V
  const _Float16* W = (which == 0) ? Wq : ((which == 1) ? Wk : Wv);
  int n0 = head * DH_;

  AccF acc[4];
  #pragma unroll
  for (int j = 0; j < 4; ++j)
    #pragma unroll
    for (int r = 0; r < 8; ++r) acc[j].f[r] = 0.0f;

  for (int k0 = 0; k0 < D_; k0 += 32) {
    v16h a = load_a_frag(xh, D_, mt, k0);
    #pragma unroll
    for (int j = 0; j < 4; ++j) {
      v16h b = load_bt_frag(W, D_, n0 + j * 16, k0);   // y = x @ W.T
      acc[j].v = wmma_f16(a, b, acc[j].v);
    }
  }

  int mloc = (lane < 16) ? 0 : 8;
  int ncol = lane & 15;

  if (which < 2) {
    // RoPE: out[d] = x[d]*cos - x[d+32]*sin ; out[d+32] = x[d+32]*cos + x[d]*sin
    // d in acc[j] (j=0,1) pairs with d+32 in acc[j+2] at same lane/VGPR.
    #pragma unroll
    for (int j = 0; j < 2; ++j) {
      #pragma unroll
      for (int r = 0; r < 8; ++r) {
        int m   = mt + r + mloc;
        int pos = m & (N_ - 1);
        int d   = j * 16 + ncol;                       // < 32
        // inv_freq = 10000^(-d/32) = exp(-d * ln(10000)/32)
        float inv = __expf(-(float)d * (9.210340371976184f / 32.0f));
        float ang = (float)pos * inv;
        float s = __sinf(ang), c = __cosf(ang);
        float x1 = acc[j].f[r], x2 = acc[j + 2].f[r];
        acc[j].f[r]     = x1 * c - x2 * s;
        acc[j + 2].f[r] = x2 * c + x1 * s;
      }
    }
    _Float16* out = (which == 0) ? Qh : Kh;
    #pragma unroll
    for (int j = 0; j < 4; ++j)
      #pragma unroll
      for (int r = 0; r < 8; ++r) {
        int m = mt + r + mloc;
        int b = m >> 11;                 // / N_
        int pos = m & (N_ - 1);
        int d = j * 16 + ncol;
        out[(((size_t)b * H_ + head) * N_ + pos) * DH_ + d] = (_Float16)acc[j].f[r];
      }
  } else {
    // V stored transposed: Vt[b,h,d,pos] so P·V B-frags read contiguously.
    #pragma unroll
    for (int j = 0; j < 4; ++j)
      #pragma unroll
      for (int r = 0; r < 8; ++r) {
        int m = mt + r + mloc;
        int b = m >> 11;
        int pos = m & (N_ - 1);
        int d = j * 16 + ncol;
        Vt[(((size_t)b * H_ + head) * DH_ + d) * N_ + pos] = (_Float16)acc[j].f[r];
      }
  }
}

// ---------------------------------------------------------------------------
// Stage 2: flash attention per (b,h). grid: (N/128, B*H)  block: 256.
// Each wave: 16 query rows, streams 32 keys/iter, online softmax in f32.
// ---------------------------------------------------------------------------
__global__ __launch_bounds__(256)
void attn(const _Float16* __restrict__ Qh,
          const _Float16* __restrict__ Kh,
          const _Float16* __restrict__ Vt,
          _Float16* __restrict__ Oh) {     // [B,N,D]
  __shared__ _Float16 lds[8 * 16 * 32];    // 1KB per wave P-tile staging
  int wave = threadIdx.x >> 5;
  int lane = threadIdx.x & 31;
  int bh = blockIdx.y;
  int b = bh >> 4, h = bh & 15;
  int q0 = blockIdx.x * 128 + wave * 16;

  const _Float16* Qb = Qh + (size_t)bh * N_ * DH_;
  const _Float16* Kb = Kh + (size_t)bh * N_ * DH_;
  const _Float16* Vb = Vt + (size_t)bh * DH_ * N_;
  _Float16* pw = lds + wave * 16 * 32;

  v16h qf0 = load_a_frag(Qb, DH_, q0, 0);
  v16h qf1 = load_a_frag(Qb, DH_, q0, 32);

  AccF o[4];
  float m_run[8], l_run[8];
  #pragma unroll
  for (int j = 0; j < 4; ++j)
    #pragma unroll
    for (int r = 0; r < 8; ++r) o[j].f[r] = 0.0f;
  #pragma unroll
  for (int r = 0; r < 8; ++r) { m_run[r] = -3.0e38f; l_run[r] = 0.0f; }

  int mloc = (lane < 16) ? 0 : 8;
  int ncol = lane & 15;
  const float scale = 0.125f;   // dh^-0.5

  for (int key0 = 0; key0 < N_; key0 += 32) {
    // S tile 16x32 = Q(16x64) . K^T
    AccF s0, s1;
    #pragma unroll
    for (int r = 0; r < 8; ++r) { s0.f[r] = 0.0f; s1.f[r] = 0.0f; }
    s0.v = wmma_f16(qf0, load_bt_frag(Kb, DH_, key0,      0), s0.v);
    s0.v = wmma_f16(qf1, load_bt_frag(Kb, DH_, key0,     32), s0.v);
    s1.v = wmma_f16(qf0, load_bt_frag(Kb, DH_, key0 + 16, 0), s1.v);
    s1.v = wmma_f16(qf1, load_bt_frag(Kb, DH_, key0 + 16,32), s1.v);

    // online softmax; row lives across a 16-lane group -> xor-reduce 1,2,4,8
    float corr[8];
    #pragma unroll
    for (int r = 0; r < 8; ++r) {
      float a = s0.f[r] * scale;
      float c2 = s1.f[r] * scale;
      float mx = fmaxf(a, c2);
      #pragma unroll
      for (int sft = 1; sft < 16; sft <<= 1) mx = fmaxf(mx, __shfl_xor(mx, sft, 32));
      float mnew = fmaxf(m_run[r], mx);
      float cr = __expf(m_run[r] - mnew);
      float p0 = __expf(a  - mnew);
      float p1 = __expf(c2 - mnew);
      float rs = p0 + p1;
      #pragma unroll
      for (int sft = 1; sft < 16; sft <<= 1) rs += __shfl_xor(rs, sft, 32);
      l_run[r] = l_run[r] * cr + rs;
      m_run[r] = mnew;
      corr[r] = cr;
      int mr = r + mloc;
      pw[mr * 32 + ncol]      = (_Float16)p0;   // C-layout -> LDS (row-major)
      pw[mr * 32 + ncol + 16] = (_Float16)p1;
    }
    #pragma unroll
    for (int j = 0; j < 4; ++j)
      #pragma unroll
      for (int r = 0; r < 8; ++r) o[j].f[r] *= corr[r];

    __asm__ volatile("" ::: "memory");       // order DS store -> DS load (in-order per wave)
    v16h pf = load_a_frag(pw, 32, 0, 0);     // P as A-matrix 16x32 from LDS
    #pragma unroll
    for (int j = 0; j < 4; ++j) {
      v16h vf = load_bt_frag(Vb, N_, j * 16, key0);   // V^T: contiguous along keys
      o[j].v = wmma_f16(pf, vf, o[j].v);
    }
    __asm__ volatile("" ::: "memory");       // keep next iter's stores after this load
  }

  #pragma unroll
  for (int r = 0; r < 8; ++r) l_run[r] = 1.0f / l_run[r];
  #pragma unroll
  for (int j = 0; j < 4; ++j)
    #pragma unroll
    for (int r = 0; r < 8; ++r) {
      int q = q0 + r + mloc;
      int d = j * 16 + ncol;
      Oh[((size_t)b * N_ + q) * D_ + h * DH_ + d] = (_Float16)(o[j].f[r] * l_run[r]);
    }
}

// ---------------------------------------------------------------------------
// Stage 3: output projection out = O @ Wo^T + bo (f32 store).
// grid: (BN/128, D/64)  block: 256
// ---------------------------------------------------------------------------
__global__ __launch_bounds__(256)
void out_proj(const _Float16* __restrict__ Oh,
              const _Float16* __restrict__ Wo,
              const float* __restrict__ bo,
              float* __restrict__ out) {
  int wave = threadIdx.x >> 5;
  int lane = threadIdx.x & 31;
  int mt = blockIdx.x * 128 + wave * 16;
  int n0 = blockIdx.y * 64;

  AccF acc[4];
  #pragma unroll
  for (int j = 0; j < 4; ++j)
    #pragma unroll
    for (int r = 0; r < 8; ++r) acc[j].f[r] = 0.0f;

  for (int k0 = 0; k0 < D_; k0 += 32) {
    v16h a = load_a_frag(Oh, D_, mt, k0);
    #pragma unroll
    for (int j = 0; j < 4; ++j) {
      v16h b = load_bt_frag(Wo, D_, n0 + j * 16, k0);
      acc[j].v = wmma_f16(a, b, acc[j].v);
    }
  }

  int mloc = (lane < 16) ? 0 : 8;
  int ncol = lane & 15;
  #pragma unroll
  for (int j = 0; j < 4; ++j)
    #pragma unroll
    for (int r = 0; r < 8; ++r) {
      int m = mt + r + mloc;
      int n = n0 + j * 16 + ncol;
      out[(size_t)m * D_ + n] = acc[j].f[r] + bo[n];
    }
}

// ---------------------------------------------------------------------------
// Host-side launch
// ---------------------------------------------------------------------------
extern "C" void kernel_launch(void* const* d_in, const int* in_sizes, int n_in,
                              void* d_out, int out_size, void* d_ws, size_t ws_size,
                              hipStream_t stream) {
  const float* x  = (const float*)d_in[0];
  const float* Wq = (const float*)d_in[1];
  const float* Wk = (const float*)d_in[2];
  const float* Wv = (const float*)d_in[3];
  const float* Wo = (const float*)d_in[4];
  const float* bo = (const float*)d_in[5];
  float* out = (float*)d_out;

  char* ws = (char*)d_ws;
  const size_t MB = 1u << 20;
  _Float16* xh  = (_Float16*)(ws + 0 * MB);   // 8 MB  [BN, D]
  _Float16* Wqh = (_Float16*)(ws + 8 * MB);   // 2 MB
  _Float16* Wkh = (_Float16*)(ws + 10 * MB);  // 2 MB
  _Float16* Wvh = (_Float16*)(ws + 12 * MB);  // 2 MB
  _Float16* Woh = (_Float16*)(ws + 14 * MB);  // 2 MB
  _Float16* Qh  = (_Float16*)(ws + 16 * MB);  // 8 MB  [B,H,N,dh]
  _Float16* Kh  = (_Float16*)(ws + 24 * MB);  // 8 MB  [B,H,N,dh]
  _Float16* Vt  = (_Float16*)(ws + 32 * MB);  // 8 MB  [B,H,dh,N]
  _Float16* Oh  = (_Float16*)(ws + 40 * MB);  // 8 MB  [B,N,D]   (48 MB total)

  // Stage 0: downconvert inputs
  cvt_f32_f16<<<1024, 256, 0, stream>>>(x,  xh,  BN_ * D_);
  cvt_f32_f16<<<512,  256, 0, stream>>>(Wq, Wqh, D_ * D_);
  cvt_f32_f16<<<512,  256, 0, stream>>>(Wk, Wkh, D_ * D_);
  cvt_f32_f16<<<512,  256, 0, stream>>>(Wv, Wvh, D_ * D_);
  cvt_f32_f16<<<512,  256, 0, stream>>>(Wo, Woh, D_ * D_);

  // Stage 1: QKV projection + RoPE + V transpose
  qkv_proj<<<dim3(BN_ / 128, H_, 3), 256, 0, stream>>>(xh, Wqh, Wkh, Wvh, Qh, Kh, Vt);

  // Stage 2: attention
  attn<<<dim3(N_ / 128, B_ * H_), 256, 0, stream>>>(Qh, Kh, Vt, Oh);

  // Stage 3: output projection + bias
  out_proj<<<dim3(BN_ / 128, D_ / 64), 256, 0, stream>>>(Oh, Woh, bo, out);
}
